// TransformerXL_88149908783419
// MI455X (gfx1250) — matrix operand order repaired and checked by reading, not compiled, not run on hardware
//
// TransformerXL layer forward for MI455X (gfx1250), bf16 WMMA + fp32 accumulate.
// Compute-bound (~140 GFLOP/call) => all GEMMs on v_wmma_f32_16x16x32_bf16.
// GEMM staging uses the CDNA5 Tensor Data Mover (tensor_load_to_lds, 6-arg
// clang-23 builtin) with double-buffered LDS and TENSORcnt-based overlap; the
// D# descriptor applies the 40-half padded LDS row stride in hardware
// (pad_interval=16DW, pad=4DW). Attention is fused flash-style; rel_shift
// handled exactly via 3 per-step BD WMMA tiles spilled through LDS.
// Workspace requirement: ~166 MB (carved with reuse below).

#include <hip/hip_runtime.h>
#include <hip/hip_bf16.h>

typedef __bf16 bf16;
typedef __attribute__((ext_vector_type(8)))  bf16  v8bf;
typedef __attribute__((ext_vector_type(16))) bf16  v16bf;
typedef __attribute__((ext_vector_type(8)))  float v8f;
typedef __attribute__((ext_vector_type(4)))  unsigned v4u;
typedef __attribute__((ext_vector_type(8)))  int   v8i;
typedef __attribute__((ext_vector_type(4)))  int   v4i;

union V16 { v16bf v; struct { v8bf lo, hi; } p; };

constexpr int B_  = 4;
constexpr int Q_  = 1024;
constexpr int D_  = 1024;
constexpr int NH  = 16;
constexpr int DH  = 64;
constexpr int DI  = 4096;
constexpr int M_  = B_ * Q_;   // 4096 token rows

__device__ inline v8f wmma_bf16(v16bf a, v16bf b, v8f c) {
  return __builtin_amdgcn_wmma_f32_16x16x32_bf16(
      /*neg_a=*/false, a, /*neg_b=*/false, b,
      /*c_mod=*/(short)0, c, /*reuse_a=*/false, /*reuse_b=*/false);
}

// xor-shuffle within 16-lane halves of the wave32 (ds_swizzle group-of-32:
// offset[14:10]=xor, [9:5]=or, [4:0]=and; and=0x1f keeps it in-wave).
template <int MASK>
__device__ inline float xor16f(float v) {
  int i = __builtin_amdgcn_ds_swizzle(__float_as_int(v), (MASK << 10) | 0x1f);
  return __int_as_float(i);
}

// --------------------------------------------------------------- TDM stage
// One descriptor moves a [128 rows x 32 halves] bf16 tile from global
// (row stride = K elements) into LDS with an 8-half pad after each row
// (=> padded LDS row stride of 40 halves, matching sA/sW below).
// D# layout per CDNA5 ISA §8.3/§8.4 (2-D tensor, groups 2/3 zero).
__device__ inline void tdm_load_tile(const bf16* gaddr, unsigned lds_off, int K) {
  const unsigned long long ga = (unsigned long long)(uintptr_t)gaddr;
  v4u g0;
  g0[0] = 1u;                                           // count=1, user mode
  g0[1] = lds_off;                                      // lds_addr (bytes)
  g0[2] = (unsigned)(ga & 0xFFFFFFFFu);                 // global_addr lo
  g0[3] = (unsigned)((ga >> 32) & 0x01FFFFFFu)          // global_addr hi (57b)
        | (2u << 30);                                   // type = 2 ("image")
  const unsigned uk = (unsigned)K;
  const unsigned TD1 = 1u << 20;                        // tensor_dim1 (rows, generous)
  v8i g1;
  g1[0] = (int)((1u << 16)      // data_size = 2 bytes
              | (1u << 20)      // pad_enable
              | (3u << 22)      // pad_interval = 16 DWORDs (one 32-half row)
              | (3u << 25));    // pad_amount   = 4 DWORDs (8 halves)
  g1[1] = (int)((uk & 0xFFFFu) << 16);                  // tensor_dim0 lo16
  g1[2] = (int)((uk >> 16) | ((TD1 & 0xFFFFu) << 16));  // dim0 hi16 | dim1 lo16
  g1[3] = (int)((TD1 >> 16) | (32u << 16));             // dim1 hi16 | tile_dim0=32
  g1[4] = 128;                                          // tile_dim1=128, tile_dim2=0
  g1[5] = (int)uk;                                      // tensor_dim0_stride lo32
  g1[6] = 0;                                            // stride hi | dim1_stride lo
  g1[7] = 0;
  const v4i z4 = {0, 0, 0, 0};
  const v8i z8 = {0, 0, 0, 0, 0, 0, 0, 0};
  __builtin_amdgcn_tensor_load_to_lds(g0, g1, z4, z4, z8, 0);
}

// ---------------------------------------------------------------- convert
__global__ void k_f32_to_bf16(const float* __restrict__ x, bf16* __restrict__ y,
                              int n) {
  int i = blockIdx.x * blockDim.x + threadIdx.x;
  if (i < n) y[i] = (bf16)x[i];
}

// ---------------------------------------------------------------- GEMM
// C[M,N] = A[M,K](bf16, row-major) x W[N,K](bf16, row-major)^T (+bias)(relu)(+resid)
// 256 threads = 8 waves (2 M x 4 N); block tile 128x128; K step 32.
// Double-buffered LDS tiles filled by the Tensor Data Mover (wave 0 issues,
// TENSORcnt overlap: DMA of tile k+1 runs under the WMMAs of tile k).
enum { EP_BIAS = 1, EP_RELU = 2, EP_RESID = 4, EP_F32 = 8, EP_BF16 = 16 };

__global__ __launch_bounds__(256) void k_gemm_bf16(
    const bf16* __restrict__ A, const bf16* __restrict__ W,
    float* __restrict__ Cf, bf16* __restrict__ Cb,
    const float* __restrict__ bias, const float* __restrict__ resid,
    int K, int N, int flags) {
  __shared__ bf16 sA[2][128][40];  // 40-half padded stride (80B, 16B aligned)
  __shared__ bf16 sW[2][128][40];
  const int tid  = threadIdx.x;
  const int lane = tid & 31, ln = lane & 15, hi = lane >> 4;
  const int wv = tid >> 5, wvM = wv >> 2, wvN = wv & 3;
  const int m0 = blockIdx.y * 128, n0 = blockIdx.x * 128;

  const unsigned ldsA0 = (unsigned)(uintptr_t)&sA[0][0][0];
  const unsigned ldsW0 = (unsigned)(uintptr_t)&sW[0][0][0];
  const unsigned BUFB  = 128u * 40u * 2u;  // LDS bytes per buffer

  v8f acc[4][2] = {};
  const int kb = hi * 16;

  if (wv == 0) {
    tdm_load_tile(A + (size_t)m0 * K, ldsA0, K);
    tdm_load_tile(W + (size_t)n0 * K, ldsW0, K);
  }

  int p = 0;
  for (int k0 = 0; k0 < K; k0 += 32) {
    const bool more = (k0 + 32) < K;
    if (wv == 0) {
      if (more) {
        const unsigned pb = (p ^ 1) ? BUFB : 0u;
        tdm_load_tile(A + (size_t)m0 * K + (k0 + 32), ldsA0 + pb, K);
        tdm_load_tile(W + (size_t)n0 * K + (k0 + 32), ldsW0 + pb, K);
        __builtin_amdgcn_s_wait_tensorcnt(2);  // buffer p ready, next in flight
      } else {
        __builtin_amdgcn_s_wait_tensorcnt(0);
      }
    }
    __syncthreads();  // buffer p visible to all waves

    V16 aF[4];
    v16bf wF[2];
#pragma unroll
    for (int mt = 0; mt < 4; ++mt) {
      const bf16* rp = &sA[p][wvM * 64 + mt * 16 + ln][0];
      aF[mt].p.lo = *(const v8bf*)(rp + kb);
      aF[mt].p.hi = *(const v8bf*)(rp + kb + 8);
    }
#pragma unroll
    for (int nt = 0; nt < 2; ++nt)
      wF[nt] = *(const v16bf*)&sW[p][wvN * 32 + nt * 16 + ln][hi * 16];
#pragma unroll
    for (int mt = 0; mt < 4; ++mt)
#pragma unroll
      for (int nt = 0; nt < 2; ++nt)
        acc[mt][nt] = wmma_bf16(aF[mt].v, wF[nt], acc[mt][nt]);

    __syncthreads();  // all reads of buffer p done before DMA overwrites it
    p ^= 1;
  }

#pragma unroll
  for (int mt = 0; mt < 4; ++mt) {
#pragma unroll
    for (int nt = 0; nt < 2; ++nt) {
      const int col  = n0 + wvN * 32 + nt * 16 + ln;
      const int rowb = m0 + wvM * 64 + mt * 16 + hi * 8;
      const float bv = (flags & EP_BIAS) ? bias[col] : 0.f;
#pragma unroll
      for (int r = 0; r < 8; ++r) {
        const size_t idx = (size_t)(rowb + r) * N + col;
        float v = acc[mt][nt][r] + bv;
        if (flags & EP_RELU)  v = fmaxf(v, 0.f);
        if (flags & EP_RESID) v += resid[idx];
        if (flags & EP_F32)   Cf[idx] = v;
        if (flags & EP_BF16)  Cb[idx] = (bf16)v;
      }
    }
  }
}

// -------------------------------------------------- QKV scatter / layouts
// heads f32 [B,Q,3D]  ->  qc,qp,k bf16 [B,N,Q,DH] ; vT bf16 [B,N,DH,Q]
__global__ void k_scatter_qkv(const float* __restrict__ heads,
                              const float* __restrict__ rwb,
                              const float* __restrict__ rrb,
                              bf16* __restrict__ qc, bf16* __restrict__ qp,
                              bf16* __restrict__ kk, bf16* __restrict__ vt) {
  int idx = blockIdx.x * 256 + threadIdx.x;          // over B*Q*NH*DH = 4M
  if (idx >= B_ * Q_ * NH * DH) return;
  const int d = idx & 63;
  const int h = (idx >> 6) & 15;
  const int q = (idx >> 10) & 1023;
  const int b = idx >> 20;
  const float* hb = heads + (size_t)(b * Q_ + q) * (3 * D_);
  const int col = h * DH + d;
  const float qv = hb[col], kv = hb[D_ + col], vv = hb[2 * D_ + col];
  const size_t hd = ((size_t)(b * NH + h) * Q_ + q) * DH + d;
  qc[hd] = (bf16)(qv + rwb[col]);
  qp[hd] = (bf16)(qv + rrb[col]);
  kk[hd] = (bf16)kv;
  vt[((size_t)(b * NH + h) * DH + d) * Q_ + q] = (bf16)vv;
}

// rkraw f32 [Q,D] -> rk bf16 [N,Q,DH]
__global__ void k_scatter_rk(const float* __restrict__ rkraw,
                             bf16* __restrict__ rk) {
  int idx = blockIdx.x * 256 + threadIdx.x;          // over NH*Q*DH = 1M
  if (idx >= NH * Q_ * DH) return;
  const int d = idx & 63;
  const int q = (idx >> 6) & 1023;
  const int h = idx >> 16;
  rk[idx] = (bf16)rkraw[(size_t)q * D_ + h * DH + d];
}

// ---------------------------------------------------------------- LayerNorm
__global__ __launch_bounds__(256) void k_layernorm(
    const float* __restrict__ x, const float* __restrict__ g,
    const float* __restrict__ b, float* __restrict__ yf,
    bf16* __restrict__ yb, int writeB16) {
  __shared__ float red[2][256];
  const int row = blockIdx.x, t = threadIdx.x;
  const float* xr = x + (size_t)row * D_;
  float s = 0.f, ss = 0.f, v4[4];
#pragma unroll
  for (int i = 0; i < 4; ++i) {
    float v = xr[t + i * 256];
    v4[i] = v; s += v; ss += v * v;
  }
  red[0][t] = s; red[1][t] = ss;
  __syncthreads();
  for (int o = 128; o > 0; o >>= 1) {
    if (t < o) { red[0][t] += red[0][t + o]; red[1][t] += red[1][t + o]; }
    __syncthreads();
  }
  const float mean = red[0][0] * (1.f / D_);
  const float var  = red[1][0] * (1.f / D_) - mean * mean;
  const float inv  = rsqrtf(var + 1e-5f);
#pragma unroll
  for (int i = 0; i < 4; ++i) {
    const int c = t + i * 256;
    const float o = (v4[i] - mean) * inv * g[c] + b[c];
    yf[(size_t)row * D_ + c] = o;
    if (writeB16) yb[(size_t)row * D_ + c] = (bf16)o;
  }
}

// ------------------------------------------------------- fused attention
// grid (Q/64, NH, B), 128 threads = 4 waves; wave owns 16 query rows.
// S = Qc*K^T (WMMA) + rel-shifted BD (3 WMMA tiles, LDS-gathered), causal mask,
// online softmax, O += P*V^T (WMMA). Exact: BD[i,j] = Qp_i . rk[j-i+Q-1].
__global__ __launch_bounds__(128) void k_attention(
    const bf16* __restrict__ qc, const bf16* __restrict__ qp,
    const bf16* __restrict__ kmat, const bf16* __restrict__ vt,
    const bf16* __restrict__ rk, bf16* __restrict__ av) {
  __shared__ float sBD[4][16][49];   // per-wave raw BD stripe (cols m-base 0..47)
  __shared__ bf16  sP[4][16][40];    // per-wave P tile for C->A relayout
  const int tid  = threadIdx.x;
  const int wv   = tid >> 5, lane = tid & 31, ln = lane & 15, hi = lane >> 4;
  const int b = blockIdx.z, h = blockIdx.y;
  const int i0 = blockIdx.x * 64 + wv * 16;
  const int kb = hi * 16;
  const size_t bh = (size_t)(b * NH + h);
  const bf16* qcb = qc + (bh * Q_ + i0) * DH;
  const bf16* qpb = qp + (bh * Q_ + i0) * DH;
  const bf16* kbv = kmat + bh * Q_ * DH;
  const bf16* vbv = vt + bh * DH * Q_;
  const bf16* rbv = rk + (size_t)h * Q_ * DH;

  V16 qcF[2], qpF[2];
#pragma unroll
  for (int t = 0; t < 2; ++t) {
    const bf16* rp  = qcb + (size_t)ln * DH + t * 32;
    const bf16* rp2 = qpb + (size_t)ln * DH + t * 32;
    qcF[t].p.lo = *(const v8bf*)(rp + kb);
    qcF[t].p.hi = *(const v8bf*)(rp + kb + 8);
    qpF[t].p.lo = *(const v8bf*)(rp2 + kb);
    qpF[t].p.hi = *(const v8bf*)(rp2 + kb + 8);
  }

  float mr[8], lrow[8];
  v8f o0 = {}, o1 = {}, o2 = {}, o3 = {};
#pragma unroll
  for (int r = 0; r < 8; ++r) { mr[r] = -1e30f; lrow[r] = 0.f; }

  for (int j0 = 0; j0 <= i0 + 15; j0 += 32) {
    // prefetch next j-step's K/V rows while this step computes
    if (j0 + 32 <= i0 + 15) {
      __builtin_prefetch(kbv + (size_t)(j0 + 32 + ln) * DH, 0, 1);
      __builtin_prefetch(vbv + (size_t)ln * Q_ + j0 + 32, 0, 1);
    }
    // --- content scores: S(16x32) = Qc(16x64) * K^T
    v8f s0 = {}, s1 = {};
#pragma unroll
    for (int t = 0; t < 2; ++t) {
      const int ko = t * 32 + hi * 16;
      v16bf w0 = *(const v16bf*)(kbv + (size_t)(j0 + ln) * DH + ko);
      v16bf w1 = *(const v16bf*)(kbv + (size_t)(j0 + 16 + ln) * DH + ko);
      s0 = wmma_bf16(qcF[t].v, w0, s0);
      s1 = wmma_bf16(qcF[t].v, w1, s1);
    }
    // --- position scores: needed BDraw columns m in [base, base+47]
    const int base = j0 - i0 + Q_ - 16;
#pragma unroll
    for (int bt = 0; bt < 3; ++bt) {
      v8f bd = {};
      int rrow = base + bt * 16 + ln;
      rrow = rrow > (Q_ - 1) ? (Q_ - 1) : rrow;   // OOB cols are masked later
#pragma unroll
      for (int t = 0; t < 2; ++t) {
        const int ko = t * 32 + hi * 16;
        v16bf wf = *(const v16bf*)(rbv + (size_t)rrow * DH + ko);
        bd = wmma_bf16(qpF[t].v, wf, bd);
      }
#pragma unroll
      for (int r = 0; r < 8; ++r) sBD[wv][hi * 8 + r][bt * 16 + ln] = bd[r];
    }
    asm volatile("s_wait_dscnt 0x0" ::: "memory");   // intra-wave LDS exchange
    __builtin_amdgcn_wave_barrier();

    // --- assemble, mask, online softmax (row = VGPR across 16-lane half)
    float fac[8];
#pragma unroll
    for (int r = 0; r < 8; ++r) {
      const int ii = hi * 8 + r;
      const int irow = i0 + ii;
      float sa = (s0[r] + sBD[wv][ii][15 + ln - ii]) * 0.125f;  // 1/sqrt(64)
      float sb = (s1[r] + sBD[wv][ii][31 + ln - ii]) * 0.125f;
      sa = (j0 + ln      <= irow) ? sa : -1e30f;
      sb = (j0 + 16 + ln <= irow) ? sb : -1e30f;
      float t = fmaxf(sa, sb);
      t = fmaxf(t, xor16f<1>(t)); t = fmaxf(t, xor16f<2>(t));
      t = fmaxf(t, xor16f<4>(t)); t = fmaxf(t, xor16f<8>(t));
      const float mn = fmaxf(mr[r], t);
      const float f  = __expf(mr[r] - mn);
      mr[r] = mn; fac[r] = f;
      const float pa = __expf(sa - mn), pb = __expf(sb - mn);
      float rs = pa + pb;
      rs += xor16f<1>(rs); rs += xor16f<2>(rs);
      rs += xor16f<4>(rs); rs += xor16f<8>(rs);
      lrow[r] = lrow[r] * f + rs;
      o0[r] *= f; o1[r] *= f; o2[r] *= f; o3[r] *= f;
      sP[wv][ii][ln]      = (bf16)pa;
      sP[wv][ii][16 + ln] = (bf16)pb;
    }
    asm volatile("s_wait_dscnt 0x0" ::: "memory");
    __builtin_amdgcn_wave_barrier();

    // --- O(16x64) += P(16x32) * V^T, K = this 32-wide j-step
    V16 pF;
    pF.p.lo = *(const v8bf*)&sP[wv][ln][kb];
    pF.p.hi = *(const v8bf*)&sP[wv][ln][kb + 8];
    const int ko2 = hi * 16;
    v16bf vf0 = *(const v16bf*)(vbv + (size_t)(ln)      * Q_ + j0 + ko2);
    v16bf vf1 = *(const v16bf*)(vbv + (size_t)(16 + ln) * Q_ + j0 + ko2);
    v16bf vf2 = *(const v16bf*)(vbv + (size_t)(32 + ln) * Q_ + j0 + ko2);
    v16bf vf3 = *(const v16bf*)(vbv + (size_t)(48 + ln) * Q_ + j0 + ko2);
    o0 = wmma_bf16(pF.v, vf0, o0);
    o1 = wmma_bf16(pF.v, vf1, o1);
    o2 = wmma_bf16(pF.v, vf2, o2);
    o3 = wmma_bf16(pF.v, vf3, o3);
  }

  // --- finalize: av bf16 [B,Q,D] head-major column packing (matches reshape)
  bf16* outp = av + ((size_t)b * Q_ + i0) * D_ + h * DH;
#pragma unroll
  for (int r = 0; r < 8; ++r) {
    const int ii = hi * 8 + r;
    const float inv = 1.f / lrow[r];
    bf16* op = outp + (size_t)ii * D_;
    op[ln]      = (bf16)(o0[r] * inv);
    op[16 + ln] = (bf16)(o1[r] * inv);
    op[32 + ln] = (bf16)(o2[r] * inv);
    op[48 + ln] = (bf16)(o3[r] * inv);
  }
}

// ================================================================ driver
extern "C" void kernel_launch(void* const* d_in, const int* in_sizes, int n_in,
                              void* d_out, int out_size, void* d_ws,
                              size_t ws_size, hipStream_t stream) {
  (void)in_sizes; (void)n_in; (void)out_size; (void)ws_size;
  const float* w    = (const float*)d_in[0];
  const float* r    = (const float*)d_in[1];
  /* d_in[2] attention_mask: causal, computed analytically */
  const float* qkvw = (const float*)d_in[3];
  const float* rw   = (const float*)d_in[4];
  const float* ow   = (const float*)d_in[5];
  const float* rwb  = (const float*)d_in[6];
  const float* rrb  = (const float*)d_in[7];
  const float* ln1g = (const float*)d_in[8];
  const float* ln1b = (const float*)d_in[9];
  const float* ffw1 = (const float*)d_in[10];
  const float* ffb1 = (const float*)d_in[11];
  const float* ffw2 = (const float*)d_in[12];
  const float* ffb2 = (const float*)d_in[13];
  const float* ln2g = (const float*)d_in[14];
  const float* ln2b = (const float*)d_in[15];
  float* out = (float*)d_out;

  char* base = (char*)d_ws;
  size_t off = 0;
  auto carve = [&](size_t bytes) -> void* {
    void* p = base + off;
    off = (off + bytes + 255) & ~(size_t)255;
    return p;
  };
  // persistent bf16 copies
  bf16* w16   = (bf16*)carve((size_t)M_ * D_ * 2);
  bf16* qw16  = (bf16*)carve((size_t)3 * D_ * D_ * 2);
  bf16* r16   = (bf16*)carve((size_t)Q_ * D_ * 2);
  bf16* rw16  = (bf16*)carve((size_t)D_ * D_ * 2);
  bf16* ow16  = (bf16*)carve((size_t)D_ * D_ * 2);
  bf16* fw116 = (bf16*)carve((size_t)DI * D_ * 2);
  bf16* fw216 = (bf16*)carve((size_t)D_ * DI * 2);
  // attention operands + activations
  bf16* qc16  = (bf16*)carve((size_t)M_ * D_ * 2);
  bf16* qp16  = (bf16*)carve((size_t)M_ * D_ * 2);
  bf16* k16   = (bf16*)carve((size_t)M_ * D_ * 2);
  bf16* vt16  = (bf16*)carve((size_t)M_ * D_ * 2);
  bf16* rk16  = (bf16*)carve((size_t)NH * Q_ * DH * 2);
  bf16* av16  = (bf16*)carve((size_t)M_ * D_ * 2);
  float* out1f = (float*)carve((size_t)M_ * D_ * 4);
  bf16*  out1b = (bf16*)carve((size_t)M_ * D_ * 2);
  // reused scratch region (64MB): heads(48M)+rkraw(4M) then tmp1/ffh/tmp2
  char* scratch = (char*)carve((size_t)64 << 20);
  float* headsf = (float*)scratch;                         // [M, 3D] f32
  float* rkraw  = (float*)(scratch + ((size_t)48 << 20));  // [Q, D] f32
  float* tmp1   = (float*)scratch;                         // [M, D] f32
  bf16*  ffh    = (bf16*)(scratch + ((size_t)16 << 20));   // [M, DI] bf16
  float* tmp2   = (float*)(scratch + ((size_t)48 << 20));  // [M, D] f32

  const dim3 blk256(256);
  auto cvt = [&](const float* src, bf16* dst, int n) {
    k_f32_to_bf16<<<dim3((n + 255) / 256), blk256, 0, stream>>>(src, dst, n);
  };
  cvt(w, w16, M_ * D_);
  cvt(qkvw, qw16, 3 * D_ * D_);
  cvt(r, r16, Q_ * D_);
  cvt(rw, rw16, D_ * D_);
  cvt(ow, ow16, D_ * D_);
  cvt(ffw1, fw116, DI * D_);
  cvt(ffw2, fw216, D_ * DI);

  // 1) heads = w @ qkv_w^T   [4096 x 3072]
  k_gemm_bf16<<<dim3(3 * D_ / 128, M_ / 128), blk256, 0, stream>>>(
      w16, qw16, headsf, nullptr, nullptr, nullptr, D_, 3 * D_, EP_F32);
  // 2) rkraw = r @ r_w^T     [1024 x 1024]
  k_gemm_bf16<<<dim3(D_ / 128, Q_ / 128), blk256, 0, stream>>>(
      r16, rw16, rkraw, nullptr, nullptr, nullptr, D_, D_, EP_F32);
  // 3) scatter into attention layouts (adds r_w_bias / r_r_bias to q)
  k_scatter_qkv<<<dim3((B_ * Q_ * NH * DH) / 256), blk256, 0, stream>>>(
      headsf, rwb, rrb, qc16, qp16, k16, vt16);
  k_scatter_rk<<<dim3((NH * Q_ * DH) / 256), blk256, 0, stream>>>(rkraw, rk16);
  // 4) fused causal attention with TransformerXL relative shift
  k_attention<<<dim3(Q_ / 64, NH, B_), dim3(128), 0, stream>>>(
      qc16, qp16, k16, vt16, rk16, av16);
  // 5) out-proj + residual(w)
  k_gemm_bf16<<<dim3(D_ / 128, M_ / 128), blk256, 0, stream>>>(
      av16, ow16, tmp1, nullptr, nullptr, w, D_, D_, EP_RESID | EP_F32);
  // 6) LN1 -> out1 (f32 residual copy + bf16 GEMM operand)
  k_layernorm<<<dim3(M_), blk256, 0, stream>>>(tmp1, ln1g, ln1b, out1f, out1b, 1);
  // 7) FFN up: relu(out1 @ ff_w1^T + b1)
  k_gemm_bf16<<<dim3(DI / 128, M_ / 128), blk256, 0, stream>>>(
      out1b, fw116, nullptr, ffh, ffb1, nullptr, D_, DI, EP_BIAS | EP_RELU | EP_BF16);
  // 8) FFN down + bias + residual(out1)
  k_gemm_bf16<<<dim3(D_ / 128, M_ / 128), blk256, 0, stream>>>(
      ffh, fw216, tmp2, nullptr, ffb2, out1f, DI, D_, EP_BIAS | EP_RESID | EP_F32);
  // 9) LN2 -> d_out (f32)
  k_layernorm<<<dim3(M_), blk256, 0, stream>>>(tmp2, ln2g, ln2b, out, nullptr, 0);
}
